// PointNetSetAbstraction_43439299232343
// MI455X (gfx1250) — matrix operand order repaired and checked by reading, not compiled, not run on hardware
//
#include <hip/hip_runtime.h>
#include <hip/hip_bf16.h>

typedef _Float16 h_t;
typedef __attribute__((ext_vector_type(8)))  _Float16 v8h;
typedef __attribute__((ext_vector_type(16))) _Float16 v16h;
typedef __attribute__((ext_vector_type(8)))  float    v8f;

#define NB      16
#define NPTS    4096
#define NPOINT  1024
#define NSAMPLE 32
#define MTOT    (NB * NPOINT * NSAMPLE)   // 524288
#define K1PAD   96                         // 67 padded to multiple of 32
#define RAD2    0.04f
#define BN_EPS  1e-5f

// ---------------------------------------------------------------- utilities
__global__ void zero_stats_kernel(float* stats, int n) {
    int i = blockIdx.x * blockDim.x + threadIdx.x;
    if (i < n) stats[i] = 0.f;
}

// Pack w0 (64x67 -> 64x96 zero-padded), w1 (64x64), w2 (128x64) into f16.
__global__ void pack_weights_kernel(const float* w0, const float* w1, const float* w2,
                                    h_t* wh0, h_t* wh1, h_t* wh2) {
    int t = blockIdx.x * blockDim.x + threadIdx.x;
    if (t < 64 * K1PAD) {
        int o = t / K1PAD, c = t % K1PAD;
        wh0[t] = (c < 67) ? (h_t)w0[o * 67 + c] : (h_t)0.f;
    } else if (t < 64 * K1PAD + 64 * 64) {
        int t2 = t - 64 * K1PAD;
        wh1[t2] = (h_t)w1[t2];
    } else if (t < 64 * K1PAD + 64 * 64 + 128 * 64) {
        int t3 = t - 64 * K1PAD - 64 * 64;
        wh2[t3] = (h_t)w2[t3];
    }
}

// ------------------------------------------------------- farthest point sample
// One block per batch; 1024 threads hold 4 points each in registers.
__global__ __launch_bounds__(1024)
void fps_kernel(const float* __restrict__ xyz, int* __restrict__ fps_idx) {
    int b = blockIdx.x;
    int t = threadIdx.x;
    const float* X = xyz + (size_t)b * 3 * NPTS;
    float px[4], py[4], pz[4], d[4];
#pragma unroll
    for (int i = 0; i < 4; ++i) {
        int j = t + i * 1024;
        px[i] = X[j]; py[i] = X[NPTS + j]; pz[i] = X[2 * NPTS + j];
        d[i] = 1e10f;
    }
    __shared__ float rv[1024];
    __shared__ int   ri[1024];
    int f = 0;
    for (int it = 0; it < NPOINT; ++it) {
        if (t == 0) fps_idx[b * NPOINT + it] = f;
        float cx = X[f], cy = X[NPTS + f], cz = X[2 * NPTS + f];
        float bv = -1.f; int bi = 0;
#pragma unroll
        for (int i = 0; i < 4; ++i) {
            float dx = px[i] - cx, dy = py[i] - cy, dz = pz[i] - cz;
            float dd = dx * dx + dy * dy + dz * dz;
            d[i] = fminf(d[i], dd);
            int j = t + i * 1024;
            if (d[i] > bv) { bv = d[i]; bi = j; }
        }
        rv[t] = bv; ri[t] = bi;
        __syncthreads();
        for (int off = 512; off > 0; off >>= 1) {
            if (t < off) {
                float ov = rv[t + off]; int oi = ri[t + off];
                if (ov > rv[t] || (ov == rv[t] && oi < ri[t])) { rv[t] = ov; ri[t] = oi; }
            }
            __syncthreads();
        }
        f = ri[0];
        __syncthreads();
    }
}

// Gather sampled centroids; writes new_xyz output head (B,3,NPOINT).
__global__ void newxyz_kernel(const float* __restrict__ xyz,
                              const int* __restrict__ fps_idx,
                              float* __restrict__ out) {
    int i = blockIdx.x * blockDim.x + threadIdx.x;
    if (i >= NB * NPOINT) return;
    int b = i >> 10, s = i & 1023;
    int j = fps_idx[i];
#pragma unroll
    for (int c = 0; c < 3; ++c)
        out[b * 3 * NPOINT + c * NPOINT + s] = xyz[(size_t)b * 3 * NPTS + c * NPTS + j];
}

// ---------------------------------------------------------------- ball query
// One wave per centroid: ordered scan, first NSAMPLE in-radius indices.
__global__ void ballquery_kernel(const float* __restrict__ xyz,
                                 const float* __restrict__ newxyz_out,
                                 int* __restrict__ gidx) {
    int w    = (blockIdx.x * blockDim.x + threadIdx.x) >> 5;
    int lane = threadIdx.x & 31;
    if (w >= NB * NPOINT) return;
    int b = w >> 10, s = w & 1023;
    float cx = newxyz_out[b * 3 * NPOINT + s];
    float cy = newxyz_out[b * 3 * NPOINT + NPOINT + s];
    float cz = newxyz_out[b * 3 * NPOINT + 2 * NPOINT + s];
    const float* X = xyz + (size_t)b * 3 * NPTS;
    int* gout = gidx + (size_t)w * NSAMPLE;
    int cnt = 0, first = -1;
    for (int j0 = 0; j0 < NPTS && cnt < NSAMPLE; j0 += 32) {
        int j = j0 + lane;
        float dx = X[j] - cx, dy = X[NPTS + j] - cy, dz = X[2 * NPTS + j] - cz;
        bool pred = (dx * dx + dy * dy + dz * dz) <= RAD2;
        unsigned mask = (unsigned)__ballot(pred);
        if (first < 0 && mask) first = j0 + (__ffs(mask) - 1);
        int pos = cnt + __popc(mask & ((1u << lane) - 1u));
        if (pred && pos < NSAMPLE) gout[pos] = j;
        cnt += __popc(mask);
    }
    if (first < 0) first = 0;
    int t = cnt < NSAMPLE ? cnt : NSAMPLE;
    if (lane >= t && lane < NSAMPLE) gout[lane] = first;   // pad with first
}

// ------------------------------------------- build f16 layer-1 input (M x 96)
__global__ void pack_input_kernel(const float* __restrict__ xyz,
                                  const float* __restrict__ points,
                                  const float* __restrict__ newxyz_out,
                                  const int* __restrict__ gidx,
                                  h_t* __restrict__ Xin) {
    size_t idx = (size_t)blockIdx.x * blockDim.x + threadIdx.x;
    if (idx >= (size_t)MTOT * K1PAD) return;
    int m = (int)(idx / K1PAD);
    int c = (int)(idx % K1PAD);
    int b = m >> 15;
    int rem = m & 32767;
    int s = rem >> 5;
    int j = gidx[m];
    float val;
    if (c < 3) {
        val = xyz[(size_t)b * 3 * NPTS + c * NPTS + j]
            - newxyz_out[b * 3 * NPOINT + c * NPOINT + s];
    } else if (c < 67) {
        val = points[(size_t)b * 64 * NPTS + (size_t)(c - 3) * NPTS + j];
    } else {
        val = 0.f;
    }
    Xin[idx] = (h_t)val;
}

// ------------------------------------------------------ WMMA GEMM + BN stats
// C[M x (NT*16)] = A[M x (32*KSTEPS)] * W^T + bias, plus per-channel sum/sum^2.
// Persistent waves: weight fragments live in registers; each wave grid-strides
// over 16-row M strips streaming only A fragments. The f16 result strip is
// staged through a per-wave LDS slice (same-wave, in-order DS => no barrier)
// and written back as fully-coalesced 128-bit global stores. BN statistics
// accumulate in registers across all strips and reduce once at the end.
template<int NT, int KSTEPS>
__global__ __launch_bounds__(128)
void gemm_wmma_stats(const h_t* __restrict__ A, const h_t* __restrict__ W,
                     const float* __restrict__ bias, h_t* __restrict__ Y,
                     float* __restrict__ gsum, float* __restrict__ gsq,
                     int ntiles_m) {
    const int Nc = NT * 16;
    const int Kc = KSTEPS * 32;
    const int lane = threadIdx.x & 31;
    const int wid  = (blockIdx.x * blockDim.x + threadIdx.x) >> 5;
    const int nwaves = (gridDim.x * blockDim.x) >> 5;

    const int ma  = lane & 15;
    const int kba = (lane >> 4) * 8;    // A: K sub-block per half-wave
    const int nb  = lane & 15;
    const int kbb = (lane >> 4) * 8;    // B: K sub-block per half-wave
    const int rowoff = (lane >> 4) * 8;

    // ---- preload ALL weight fragments into registers (shared by all strips)
    v16h bf[NT * KSTEPS];
#pragma unroll
    for (int kt = 0; kt < KSTEPS; ++kt)
#pragma unroll
        for (int nt = 0; nt < NT; ++nt)
            bf[kt * NT + nt] = __builtin_shufflevector(
                *(const v8h*)(W + (size_t)(nt * 16 + nb) * Kc + kt * 32 + kbb),
                *(const v8h*)(W + (size_t)(nt * 16 + nb) * Kc + kt * 32 + kbb + 16),
                0, 1, 2, 3, 4, 5, 6, 7, 8, 9, 10, 11, 12, 13, 14, 15);

    float bv[NT];
#pragma unroll
    for (int nt = 0; nt < NT; ++nt) bv[nt] = bias[nt * 16 + (lane & 15)];

    float ssum[NT] = {};
    float ssq[NT]  = {};

    // per-wave private staging slice (no cross-wave sharing -> no barriers)
    __shared__ __align__(16) h_t stage[4 * 16 * NT * 16];
    h_t* st = stage + (threadIdx.x >> 5) * (16 * Nc);

    for (int tile = wid; tile < ntiles_m; tile += nwaves) {
        const int mbase = tile * 16;
        // prefetch next strip's A rows (global_prefetch_b8)
        int ntile = tile + nwaves;
        if (ntile < ntiles_m)
            __builtin_prefetch(A + (size_t)(ntile * 16 + ma) * Kc, 0, 1);

        v8f acc[NT] = {};
        const h_t* arow = A + (size_t)(mbase + ma) * Kc + kba;
#pragma unroll
        for (int kt = 0; kt < KSTEPS; ++kt) {
            v8h alo = *(const v8h*)(arow + kt * 32);
            v8h ahi = *(const v8h*)(arow + kt * 32 + 16);
            v16h a = __builtin_shufflevector(alo, ahi,
                     0, 1, 2, 3, 4, 5, 6, 7, 8, 9, 10, 11, 12, 13, 14, 15);
#pragma unroll
            for (int nt = 0; nt < NT; ++nt)
                acc[nt] = __builtin_amdgcn_wmma_f32_16x16x32_f16(
                              false, a, false, bf[kt * NT + nt], (short)0,
                              acc[nt], false, false);
        }
        // epilogue: bias + BN partial sums; f16 strip into per-wave LDS slice
#pragma unroll
        for (int nt = 0; nt < NT; ++nt) {
            int col = nt * 16 + (lane & 15);
#pragma unroll
            for (int r = 0; r < 8; ++r) {
                float v = acc[nt][r] + bv[nt];
                st[(rowoff + r) * Nc + col] = (h_t)v;
                ssum[nt] += v;
                ssq[nt]  += v * v;
            }
        }
        // same-wave LDS readback -> coalesced 128-bit global stores
        // (16 full rows of a row-major matrix are one contiguous range)
        h_t* ybase = Y + (size_t)mbase * Nc;
#pragma unroll
        for (int i = 0; i < (2 * Nc) / 32; ++i) {
            int chunk = lane + i * 32;
            v8h vv = *(const v8h*)(st + chunk * 8);
            *(v8h*)(ybase + (size_t)chunk * 8) = vv;
        }
    }

    // ---- one reduction per wave: LDS per-channel, then global atomics
    __shared__ float ls[NT * 16];
    __shared__ float lq[NT * 16];
    for (int c = threadIdx.x; c < Nc; c += blockDim.x) { ls[c] = 0.f; lq[c] = 0.f; }
    __syncthreads();
#pragma unroll
    for (int nt = 0; nt < NT; ++nt) {
        int col = nt * 16 + (lane & 15);
        atomicAdd(&ls[col], ssum[nt]);
        atomicAdd(&lq[col], ssq[nt]);
    }
    __syncthreads();
    for (int c = threadIdx.x; c < Nc; c += blockDim.x) {
        atomicAdd(&gsum[c], ls[c]);
        atomicAdd(&gsq[c],  lq[c]);
    }
}

// ------------------------------------ batchnorm + relu (f16, 8-wide vectors)
__global__ void bn_relu_kernel(h_t* __restrict__ Y,
                               const float* __restrict__ gsum,
                               const float* __restrict__ gsq,
                               const float* __restrict__ g,
                               const float* __restrict__ be,
                               int Nc, size_t total8) {
    size_t i = (size_t)blockIdx.x * blockDim.x + threadIdx.x;
    if (i >= total8) return;
    size_t base = i * 8;
    int c0 = (int)(base % Nc);       // Nc multiple of 8 -> no wrap in a chunk
    const float invM = 1.f / (float)MTOT;
    v8h x = *(v8h*)(Y + base);
    v8h o;
#pragma unroll
    for (int j = 0; j < 8; ++j) {
        int c = c0 + j;
        float mean = gsum[c] * invM;
        float var  = gsq[c] * invM - mean * mean;
        float y = g[c] * ((float)x[j] - mean) * rsqrtf(var + BN_EPS) + be[c];
        o[j] = (h_t)fmaxf(y, 0.f);
    }
    *(v8h*)(Y + base) = o;
}

// ---------------------- final: BN + ReLU + max over K + transpose to (B,C,S)
// One thread per (b, s, 8-channel group): 128-bit loads per k-sample.
__global__ void bn_relu_max_kernel(const h_t* __restrict__ Y3,
                                   const float* __restrict__ gsum,
                                   const float* __restrict__ gsq,
                                   const float* __restrict__ g,
                                   const float* __restrict__ be,
                                   float* __restrict__ out) {
    int i = blockIdx.x * blockDim.x + threadIdx.x;
    if (i >= NB * NPOINT * 16) return;
    int cg = i & 15;                 // channel group (8 channels)
    int s  = (i >> 4) & 1023;
    int b  = i >> 14;
    int c0 = cg * 8;
    const float invM = 1.f / (float)MTOT;
    float mean[8], rs[8], gc[8], bc[8];
#pragma unroll
    for (int j = 0; j < 8; ++j) {
        int c = c0 + j;
        mean[j] = gsum[c] * invM;
        float var = gsq[c] * invM - mean[j] * mean[j];
        rs[j] = rsqrtf(var + BN_EPS);
        gc[j] = g[c]; bc[j] = be[c];
    }
    const h_t* base = Y3 + ((size_t)(b * NPOINT + s) * NSAMPLE) * 128 + c0;
    float best[8] = {};
#pragma unroll 4
    for (int k = 0; k < NSAMPLE; ++k) {
        v8h v = *(const v8h*)(base + (size_t)k * 128);
#pragma unroll
        for (int j = 0; j < 8; ++j) {
            float y = gc[j] * ((float)v[j] - mean[j]) * rs[j] + bc[j];
            best[j] = fmaxf(best[j], fmaxf(y, 0.f));
        }
    }
#pragma unroll
    for (int j = 0; j < 8; ++j)
        out[(size_t)b * 128 * NPOINT + (size_t)(c0 + j) * NPOINT + s] = best[j];
}

// ----------------------------------------------------------------- launcher
extern "C" void kernel_launch(void* const* d_in, const int* in_sizes, int n_in,
                              void* d_out, int out_size, void* d_ws, size_t ws_size,
                              hipStream_t stream) {
    const float* xyz    = (const float*)d_in[0];
    const float* points = (const float*)d_in[1];
    const float* w0 = (const float*)d_in[2];
    const float* b0 = (const float*)d_in[3];
    const float* g0 = (const float*)d_in[4];
    const float* be0 = (const float*)d_in[5];
    const float* w1 = (const float*)d_in[6];
    const float* b1 = (const float*)d_in[7];
    const float* g1 = (const float*)d_in[8];
    const float* be1 = (const float*)d_in[9];
    const float* w2 = (const float*)d_in[10];
    const float* b2 = (const float*)d_in[11];
    const float* g2 = (const float*)d_in[12];
    const float* be2 = (const float*)d_in[13];
    float* out = (float*)d_out;
    char* ws = (char*)d_ws;

    // workspace layout (256B aligned regions)
    size_t off = 0;
    auto take = [&](size_t bytes) { size_t o = off; off = (off + bytes + 255) & ~(size_t)255; return o; };
    int*   fps_idx = (int*)(ws + take((size_t)NB * NPOINT * 4));
    int*   gidx    = (int*)(ws + take((size_t)MTOT * 4));
    float* stats   = (float*)(ws + take(6 * 128 * 4));
    h_t*   wh0     = (h_t*)(ws + take((size_t)64 * K1PAD * 2));
    h_t*   wh1     = (h_t*)(ws + take((size_t)64 * 64 * 2));
    h_t*   wh2     = (h_t*)(ws + take((size_t)128 * 64 * 2));
    h_t*   Xin     = (h_t*)(ws + take((size_t)MTOT * K1PAD * 2));  // also reused as Y2
    h_t*   Y1      = (h_t*)(ws + take((size_t)MTOT * 64 * 2));
    h_t*   Y3      = (h_t*)(ws + take((size_t)MTOT * 128 * 2));
    h_t*   Y2      = Xin;  // layer-1 input dead after GEMM1

    float* sum1 = stats,       *sq1 = stats + 128;
    float* sum2 = stats + 256, *sq2 = stats + 384;
    float* sum3 = stats + 512, *sq3 = stats + 640;

    // 0) zero BN statistics
    zero_stats_kernel<<<3, 256, 0, stream>>>(stats, 768);
    // 1) f32 -> f16 weight packing (w0 zero-padded 67 -> 96)
    pack_weights_kernel<<<(64 * K1PAD + 64 * 64 + 128 * 64 + 255) / 256, 256, 0, stream>>>(
        w0, w1, w2, wh0, wh1, wh2);
    // 2) farthest point sampling
    fps_kernel<<<NB, 1024, 0, stream>>>(xyz, fps_idx);
    // 3) gather centroids -> output head (B,3,NPOINT)
    newxyz_kernel<<<(NB * NPOINT + 255) / 256, 256, 0, stream>>>(xyz, fps_idx, out);
    // 4) ball query (one wave per centroid)
    ballquery_kernel<<<(NB * NPOINT * 32 + 255) / 256, 256, 0, stream>>>(xyz, out, gidx);
    // 5) build grouped f16 input, 96-padded
    {
        size_t tot = (size_t)MTOT * K1PAD;
        pack_input_kernel<<<(unsigned)((tot + 255) / 256), 256, 0, stream>>>(
            xyz, points, out, gidx, Xin);
    }
    const int ntiles_m = MTOT / 16;        // 32768 strips of 16 rows
    const int gemm_blocks = 1024;          // 4096 waves -> 8 strips/wave
    const size_t n8_64 = (size_t)MTOT * 64 / 8;
    // 6) layer 1: 96 -> 64
    gemm_wmma_stats<4, 3><<<gemm_blocks, 128, 0, stream>>>(Xin, wh0, b0, Y1, sum1, sq1, ntiles_m);
    bn_relu_kernel<<<(unsigned)((n8_64 + 255) / 256), 256, 0, stream>>>(
        Y1, sum1, sq1, g0, be0, 64, n8_64);
    // 7) layer 2: 64 -> 64
    gemm_wmma_stats<4, 2><<<gemm_blocks, 128, 0, stream>>>(Y1, wh1, b1, Y2, sum2, sq2, ntiles_m);
    bn_relu_kernel<<<(unsigned)((n8_64 + 255) / 256), 256, 0, stream>>>(
        Y2, sum2, sq2, g1, be1, 64, n8_64);
    // 8) layer 3: 64 -> 128
    gemm_wmma_stats<8, 2><<<gemm_blocks, 128, 0, stream>>>(Y2, wh2, b2, Y3, sum3, sq3, ntiles_m);
    // 9) BN + ReLU + max over K=32 + transpose -> output tail (B,128,NPOINT)
    bn_relu_max_kernel<<<(NB * NPOINT * 16 + 255) / 256, 256, 0, stream>>>(
        Y3, sum3, sq3, g2, be2, out + NB * 3 * NPOINT);
}
